// RiskyObject_55465207661144
// MI455X (gfx1250) — matrix operand order repaired and checked by reading, not compiled, not run on hardware
//
#include <hip/hip_runtime.h>
#include <math.h>

// ---------------------------------------------------------------------------
// GRU tracker for MI455X (gfx1250, wave32, WMMA 16x16x32 f16 + TDM).
//  K1: convert all weights f32->f16 (once per launch, deterministic)
//  K2: phi GEMM  (T*31, 4096) @ (4096,256) + bias + relu  -> f16   [parallel]
//      A-tiles staged global->LDS via Tensor Data Mover (double buffered,
//      TENSORcnt + workgroup barrier handoff).
//  K3: gi0 GEMM  (T*30, 512) @ (512,768) + bias -> f32             [parallel]
//  K4: act/lab outputs
//  K5: persistent single-workgroup sequential kernel: 100 steps of fused
//      WMMA GRU cells through LDS (no gh materialization), cor-GRU, d1 WMMA,
//      d2 + log-softmax loss.
// ---------------------------------------------------------------------------

typedef __attribute__((ext_vector_type(16))) _Float16     v16h;
typedef __attribute__((ext_vector_type(8)))  float        v8f;
typedef __attribute__((ext_vector_type(4)))  unsigned int v4u;
typedef __attribute__((ext_vector_type(8)))  int          v8i;
typedef __attribute__((ext_vector_type(4)))  int          v4i;

constexpr int kT  = 100;
constexpr int kNB = 30;
constexpr int kH  = 256;
constexpr int kHC = 32;
constexpr int kXD = 4096;
constexpr int kN3 = 768;          // 3*H
constexpr int kPhiRows = kT * 31; // 3100
constexpr int kGiRows  = kT * kNB;// 3000

#define WMMA_F16(a, b, c) \
  __builtin_amdgcn_wmma_f32_16x16x32_f16(false, (a), false, (b), (short)0, (c), false, false)

__device__ __forceinline__ float sigmoidf_(float x) { return 1.0f / (1.0f + expf(-x)); }

// ---- WMMA fragment loaders (ISA 7.12.2 wave32 layouts) --------------------

// A (16x32, f16) from row-major f16 (LDS or global). Lane holds row m=lane&15;
// lanes 0-15: K=0-7 & 16-23, lanes 16-31: K=8-15 & 24-31, 2 halves per VGPR.
__device__ __forceinline__ v16h load_a_f16(const _Float16* base, int stride, int kt, int lane) {
  const int m  = lane & 15;
  const int kh = (lane & 16) ? 8 : 0;
  const _Float16* p = base + m * stride + kt + kh;
  v16h a;
#pragma unroll
  for (int v = 0; v < 8; ++v) {
    const int kk = ((v & 4) << 2) + ((v & 3) << 1); // 0,2,4,6,16,18,20,22
    a[2 * v]     = p[kk];
    a[2 * v + 1] = p[kk + 1];
  }
  return a;
}

// B (32x16 f16) from row-major weight w[n][k] (B[k][n] = w[n][k]).
// Lane holds col n=lane&15; lanes 0-15: K=0-7 & 16-23, lanes 16-31: K=8-15 & 24-31.
__device__ __forceinline__ v16h load_b_w(const _Float16* w, int K, int ncol, int kt, int lane) {
  const int n  = lane & 15;
  const int kb = (lane & 16) ? 8 : 0;
  const _Float16* p = w + (size_t)(ncol + n) * K + kt + kb;
  v16h b;
#pragma unroll
  for (int v = 0; v < 8; ++v) {
    const int kk = ((v & 4) << 2) + ((v & 3) << 1); // 0,2,4,6,16,18,20,22
    b[2 * v]     = p[kk];
    b[2 * v + 1] = p[kk + 1];
  }
  return b;
}

// ---- Tensor Data Mover: 2D tile (nrows x 32 f32, row stride 4096) -> LDS --
// D# per ISA ch8: group0 = {flags, lds_addr, global_addr[56:0], type=2},
// group1 = {mask/data_size, dims, tile dims, strides}.  tensor dims == tile
// dims (no OOB reliance; caller clamps nrows).  6-arg builtin (clang-23 /
// therock toolchain): groups 2/3 + trailing int32x8 unused for 2D tiles.
__device__ __forceinline__ void tdm_load_tile_f32(const float* gaddr, unsigned lds_byte_off,
                                                  int nrows) {
  const unsigned long long ga = (unsigned long long)(size_t)gaddr;
  v4u g0;
  g0[0] = 1u;                                             // count=1, user mode
  g0[1] = lds_byte_off;                                   // lds_addr (bytes)
  g0[2] = (unsigned)(ga & 0xFFFFFFFFu);                   // global_addr[31:0]
  g0[3] = (unsigned)((ga >> 32) & 0x01FFFFFFu) | 0x80000000u; // addr[56:32] | type=2
  v8i g1;
  g1[0] = 0x00020000;                 // workgroup_mask=0, data_size=2 (4B)
  g1[1] = (int)(32u << 16);           // tensor_dim0[15:0]=32
  g1[2] = (int)((unsigned)nrows << 16);        // dim0[31:16]=0 | tensor_dim1[15:0]=nrows
  g1[3] = (int)(32u << 16);           // tensor_dim1[31:16]=0 | tile_dim0=32
  g1[4] = nrows;                      // tile_dim1=nrows, tile_dim2=0
  g1[5] = kXD;                        // tensor_dim0_stride[31:0] = 4096
  g1[6] = 0;
  g1[7] = 0;
  v4i gz4 = {0, 0, 0, 0};
  v8i gz8 = {0, 0, 0, 0, 0, 0, 0, 0};
  __builtin_amdgcn_tensor_load_to_lds(g0, g1, gz4, gz4, gz8, 0);
}

// ---------------------------------------------------------------------------
__global__ void cvt_f16_kernel(const float* __restrict__ in, _Float16* __restrict__ out, int n) {
  int i = blockIdx.x * blockDim.x + threadIdx.x;
  if (i < n) out[i] = (_Float16)in[i];
}

// phi: out = relu(src @ phi_w.T + b). src (3100,4096) f32, w16 (256,4096) f16.
// Block: 128 thr (4 waves), one 16-row tile x 128 cols (2 tiles per wave).
// A tiles (16x32 f32) pulled into LDS by the TDM, double buffered.
__global__ void phi_kernel(const float* __restrict__ src, const _Float16* __restrict__ w16,
                           const float* __restrict__ bias, _Float16* __restrict__ out) {
  __shared__ float stage[2][16 * 32];       // only LDS in this kernel -> offset 0
  constexpr int NK = kXD / 32;              // 128 K-steps
  const int lane = threadIdx.x & 31;
  const int wv   = threadIdx.x >> 5;
  const int row0 = blockIdx.x * 16;
  const int nrows = (kPhiRows - row0 < 16) ? (kPhiRows - row0) : 16;
  const int col0 = blockIdx.y * 128 + wv * 32;

  v8f acc0 = {}, acc1 = {};
  if (wv == 0) tdm_load_tile_f32(src + (size_t)row0 * kXD, 0u, nrows);

  for (int ks = 0; ks < NK; ++ks) {
    const int par = ks & 1;
    if (wv == 0) {
      if (ks + 1 < NK) {
        tdm_load_tile_f32(src + (size_t)row0 * kXD + (ks + 1) * 32,
                          (unsigned)((par ^ 1) * (16 * 32 * 4)), nrows);
        __builtin_amdgcn_s_wait_tensorcnt(1);   // previous tile complete
      } else {
        __builtin_amdgcn_s_wait_tensorcnt(0);
      }
    }
    __syncthreads();
    // A fragment from staged f32 tile (16x32 row-major in LDS)
    v16h a;
    {
      const float* p = &stage[par][(lane & 15) * 32 + ((lane & 16) ? 8 : 0)];
#pragma unroll
      for (int v = 0; v < 8; ++v) {
        const int kk = ((v & 4) << 2) + ((v & 3) << 1);
        a[2 * v]     = (_Float16)p[kk];
        a[2 * v + 1] = (_Float16)p[kk + 1];
      }
    }
    const int kt = ks * 32;
    acc0 = WMMA_F16(a, load_b_w(w16, kXD, col0,      kt, lane), acc0);
    acc1 = WMMA_F16(a, load_b_w(w16, kXD, col0 + 16, kt, lane), acc1);
    __syncthreads();   // tile consumed before TDM reuses this buffer
  }

  const int n = lane & 15;
  const float b0 = bias[col0 + n];
  const float b1 = bias[col0 + 16 + n];
#pragma unroll
  for (int v = 0; v < 8; ++v) {
    const int r = row0 + v + ((lane & 16) ? 8 : 0);
    if (r < kPhiRows) {
      out[(size_t)r * kH + col0 + n]      = (_Float16)fmaxf(acc0[v] + b0, 0.0f);
      out[(size_t)r * kH + col0 + 16 + n] = (_Float16)fmaxf(acc1[v] + b1, 0.0f);
    }
  }
}

// gi0: (3000,512)@(512,768)+bih -> f32. Row (t,b): k<256 -> phi[t,b+1,k],
// k>=256 -> phi[t,0,k-256]  (x_t = concat(xv[1:], bcast(xv[0]))).
// Row index clamped (not guarded) so the K-loop stays branch-free; garbage
// rows only feed discarded D rows (WMMA rows are independent).
__global__ void gi0_kernel(const _Float16* __restrict__ phi, const _Float16* __restrict__ w16,
                           const float* __restrict__ bih, float* __restrict__ gi) {
  const int lane = threadIdx.x & 31;
  const int wv   = threadIdx.x >> 5;
  const int row0 = blockIdx.x * 16;
  const int col0 = (blockIdx.y * 4 + wv) * 16;
  const int m  = lane & 15;
  const int r  = row0 + m;
  const int rc = (r < kGiRows) ? r : (kGiRows - 1);   // clamp, keep loads in-bounds
  const int kh = (lane & 16) ? 8 : 0;
  const int tt = rc / kNB;
  const int bb = rc % kNB;
  const _Float16* rowA = phi + ((size_t)(tt * 31 + bb + 1)) * kH; // k in [0,256)
  const _Float16* rowB = phi + ((size_t)(tt * 31)) * kH;          // k in [256,512)
  v8f acc = {};
  for (int kt = 0; kt < 2 * kH; kt += 32) {
    const _Float16* p = ((kt < kH) ? (rowA + kt) : (rowB + (kt - kH))) + kh;
    v16h a;
#pragma unroll
    for (int v = 0; v < 8; ++v) {
      const int kk = ((v & 4) << 2) + ((v & 3) << 1);
      a[2 * v]     = p[kk];
      a[2 * v + 1] = p[kk + 1];
    }
    v16h b = load_b_w(w16, 2 * kH, col0, kt, lane);
    acc = WMMA_F16(a, b, acc);
  }
  const int col = col0 + (lane & 15);
  const float bv = bih[col];
#pragma unroll
  for (int v = 0; v < 8; ++v) {
    const int rr = row0 + v + ((lane & 16) ? 8 : 0);
    if (rr < kGiRows) gi[(size_t)rr * kN3 + col] = acc[v] + bv;
  }
}

__global__ void out_meta_kernel(const float* __restrict__ y, float* __restrict__ act,
                                float* __restrict__ lab) {
  int i = blockIdx.x * blockDim.x + threadIdx.x;
  if (i < kT * kNB) {
    act[i] = (y[(size_t)i * 6] != 0.0f) ? 1.0f : 0.0f;
    lab[i] = y[(size_t)i * 6 + 5];
  }
}

// ---------------------------------------------------------------------------
struct SeqArgs {
  const float* y;
  const float* gi_f0; const float* gi_m0;
  const _Float16* w_f_whh0; const _Float16* w_f_wih1; const _Float16* w_f_whh1;
  const _Float16* w_m_whh0; const _Float16* w_m_wih1; const _Float16* w_m_whh1;
  const _Float16* w_d1;
  const float* f_bhh0; const float* f_bih1; const float* f_bhh1;
  const float* m_bhh0; const float* m_bih1; const float* m_bhh1;
  const float* cor_wih; const float* cor_whh; const float* cor_bih; const float* cor_bhh;
  const float* d1_b; const float* d2_w; const float* d2_b;
  float* Hf; float* Hm; float* Hc;   // ping-pong state [2][...]
  float* out;
};

constexpr int SEQ_THREADS = 512;
constexpr int SEQ_WAVES   = SEQ_THREADS / 32;
constexpr int HA_S = 264;   // padded f16 row stride for h staging
constexpr int DA_S = 296;   // padded f16 row stride for d1 input

__global__ void __launch_bounds__(SEQ_THREADS) seq_kernel(SeqArgs A) {
  // hA rows: 0-31 hf0 | 32-63 hf1 | 64-95 hm0 | 96-127 hm1   (b in [0,30), pad 0)
  __shared__ _Float16 hA[128][HA_S];
  // hB rows: 0-31 f0 | 32-63 m0
  __shared__ _Float16 hB[64][HA_S];
  __shared__ _Float16 dA[32][DA_S];      // [m1 | c1] for d1
  __shared__ _Float16 h1s[30][128];
  __shared__ float s_hc[30][kHC];
  __shared__ float s_cor[32][4];
  __shared__ float s_log[30][2];
  __shared__ int prev_ids[32], prev_act[32];
  __shared__ int s_ids[32], s_act[32], s_lab[32], s_found[32], s_idx[32];
  __shared__ float s_loss;

  const int tid  = threadIdx.x;
  const int lane = tid & 31;
  const int wave = tid >> 5;

  if (tid < 32) { prev_ids[tid] = 0; prev_act[tid] = 0; }
  if (tid == 0) s_loss = 0.0f;
  __syncthreads();

  for (int t = 0; t < kT; ++t) {
    const int pp = t & 1, np = 1 - pp;
    const float* Hf_prev = A.Hf + (size_t)pp * kNB * 2 * kH;
    float*       Hf_next = A.Hf + (size_t)np * kNB * 2 * kH;
    const float* Hm_prev = A.Hm + (size_t)pp * kNB * 2 * kH;
    float*       Hm_next = A.Hm + (size_t)np * kNB * 2 * kH;
    const float* Hc_prev = A.Hc + (size_t)pp * kNB * kHC;
    float*       Hc_next = A.Hc + (size_t)np * kNB * kHC;

    // ---- Phase G: per-step inputs + id matching ----
    if (tid < kNB) {
      const float* yr = A.y + (size_t)(t * kNB + tid) * 6;
      const int id = (int)yr[0];
      s_ids[tid] = id;
      s_act[tid] = (id != 0);
      s_lab[tid] = (int)yr[5];
      s_cor[tid][0] = yr[1] * (1.0f / 1080.0f);
      s_cor[tid][1] = yr[2] * (1.0f / 720.0f);
      s_cor[tid][2] = yr[3] * (1.0f / 1080.0f);
      s_cor[tid][3] = yr[4] * (1.0f / 720.0f);
      int fnd = 0, ix = 0;
      for (int j = 0; j < kNB; ++j)
        if (prev_act[j] && prev_ids[j] == id) { fnd = 1; ix = j; break; }
      s_found[tid] = fnd; s_idx[tid] = ix;
    }
    __syncthreads();

    // ---- Phase S: gather prev hidden states into LDS as f16 ----
    for (int e = tid; e < 128 * kH; e += SEQ_THREADS) {
      const int row = e >> 8, k = e & (kH - 1);
      const int g = row >> 5, b = row & 31;
      float v = 0.0f;
      if (b < kNB && s_found[b]) {
        const float* Hp = (g < 2) ? Hf_prev : Hm_prev;
        v = Hp[((size_t)s_idx[b] * 2 + (g & 1)) * kH + k];
      }
      hA[row][k] = (_Float16)v;
    }
    __syncthreads();

    // ---- Phase 1: layer-0 GRUs (flow f0, main m0), fused r/z/n tiles ----
    for (int task = wave; task < 64; task += SEQ_WAVES) {
      const int br = task >> 5;           // 0=flow 1=main
      const int rt = (task >> 4) & 1;
      const int cc = task & 15;
      const _Float16* W = br ? A.w_m_whh0 : A.w_f_whh0;
      const float* gi   = br ? A.gi_m0 : A.gi_f0;
      const float* bhh  = br ? A.m_bhh0 : A.f_bhh0;
      float* Hnx        = br ? Hm_next : Hf_next;
      const int gb = br * 64;             // hA group base (layer 0)
      v8f aR = {}, aZ = {}, aN = {};
      for (int kt = 0; kt < kH; kt += 32) {
        v16h av = load_a_f16(&hA[gb + rt * 16][0], HA_S, kt, lane);
        aR = WMMA_F16(av, load_b_w(W, kH, cc * 16,          kt, lane), aR);
        aZ = WMMA_F16(av, load_b_w(W, kH, cc * 16 + kH,     kt, lane), aZ);
        aN = WMMA_F16(av, load_b_w(W, kH, cc * 16 + 2 * kH, kt, lane), aN);
      }
      const int col = cc * 16 + (lane & 15);
      const float bhr = bhh[col], bhz = bhh[col + kH], bhn = bhh[col + 2 * kH];
#pragma unroll
      for (int v = 0; v < 8; ++v) {
        const int bi = rt * 16 + v + ((lane & 16) ? 8 : 0);
        if (bi < kNB) {
          const float* gr = gi + (size_t)(t * kNB + bi) * kN3;
          const float r  = sigmoidf_(gr[col] + aR[v] + bhr);
          const float z  = sigmoidf_(gr[col + kH] + aZ[v] + bhz);
          const float nn = tanhf(gr[col + 2 * kH] + r * (aN[v] + bhn));
          const float hp = (float)hA[gb + bi][col];
          const float h  = (1.0f - z) * nn + z * hp;
          hB[br * 32 + bi][col] = (_Float16)h;
          Hnx[((size_t)bi * 2 + 0) * kH + col] = h;
        }
      }
    }
    __syncthreads();

    // ---- Phase 2: layer-1 GRUs (f1, m1): gi from layer-0 out, gh from prev ----
    for (int task = wave; task < 64; task += SEQ_WAVES) {
      const int br = task >> 5;
      const int rt = (task >> 4) & 1;
      const int cc = task & 15;
      const _Float16* Wi = br ? A.w_m_wih1 : A.w_f_wih1;
      const _Float16* Wh = br ? A.w_m_whh1 : A.w_f_whh1;
      const float* bih   = br ? A.m_bih1 : A.f_bih1;
      const float* bhh   = br ? A.m_bhh1 : A.f_bhh1;
      float* Hnx         = br ? Hm_next : Hf_next;
      const int ib = br * 32;             // hB group base (layer-0 output)
      const int gb = br * 64 + 32;        // hA group base (prev layer-1 state)
      v8f iR = {}, iZ = {}, iN = {}, hR = {}, hZ = {}, hN = {};
      for (int kt = 0; kt < kH; kt += 32) {
        v16h ai = load_a_f16(&hB[ib + rt * 16][0], HA_S, kt, lane);
        iR = WMMA_F16(ai, load_b_w(Wi, kH, cc * 16,          kt, lane), iR);
        iZ = WMMA_F16(ai, load_b_w(Wi, kH, cc * 16 + kH,     kt, lane), iZ);
        iN = WMMA_F16(ai, load_b_w(Wi, kH, cc * 16 + 2 * kH, kt, lane), iN);
        v16h ah = load_a_f16(&hA[gb + rt * 16][0], HA_S, kt, lane);
        hR = WMMA_F16(ah, load_b_w(Wh, kH, cc * 16,          kt, lane), hR);
        hZ = WMMA_F16(ah, load_b_w(Wh, kH, cc * 16 + kH,     kt, lane), hZ);
        hN = WMMA_F16(ah, load_b_w(Wh, kH, cc * 16 + 2 * kH, kt, lane), hN);
      }
      const int col = cc * 16 + (lane & 15);
      const float bir = bih[col], biz = bih[col + kH], big = bih[col + 2 * kH];
      const float bhr = bhh[col], bhz = bhh[col + kH], bhn = bhh[col + 2 * kH];
#pragma unroll
      for (int v = 0; v < 8; ++v) {
        const int bi = rt * 16 + v + ((lane & 16) ? 8 : 0);
        if (bi < kNB) {
          const float r  = sigmoidf_((iR[v] + bir) + (hR[v] + bhr));
          const float z  = sigmoidf_((iZ[v] + biz) + (hZ[v] + bhz));
          const float nn = tanhf((iN[v] + big) + r * (hN[v] + bhn));
          const float hp = (float)hA[gb + bi][col];
          const float h  = (1.0f - z) * nn + z * hp;
          Hnx[((size_t)bi * 2 + 1) * kH + col] = h;
          if (br) dA[bi][col] = (_Float16)h;   // m1 -> d1 input
        }
      }
    }
    __syncthreads();

    // ---- Phase 3: cor GRU (HC=32), scalar ----
    for (int e = tid; e < kNB * kHC; e += SEQ_THREADS) {
      const int b = e >> 5, c = e & 31;
      s_hc[b][c] = s_found[b] ? Hc_prev[s_idx[b] * kHC + c] : 0.0f;
    }
    __syncthreads();
    for (int e = tid; e < kNB * kHC; e += SEQ_THREADS) {
      const int b = e >> 5, c = e & 31;
      float gv[3], hv[3];
#pragma unroll
      for (int g = 0; g < 3; ++g) {
        const int j = c + kHC * g;
        float si = A.cor_bih[j];
        const float* wi = A.cor_wih + j * 4;
        for (int k = 0; k < 4; ++k) si += s_cor[b][k] * wi[k];
        float sh = A.cor_bhh[j];
        const float* wh = A.cor_whh + (size_t)j * kHC;
        for (int k = 0; k < kHC; ++k) sh += s_hc[b][k] * wh[k];
        gv[g] = si; hv[g] = sh;
      }
      const float r  = sigmoidf_(gv[0] + hv[0]);
      const float z  = sigmoidf_(gv[1] + hv[1]);
      const float nn = tanhf(gv[2] + r * hv[2]);
      const float h  = (1.0f - z) * nn + z * s_hc[b][c];
      dA[b][kH + c] = (_Float16)h;
      Hc_next[b * kHC + c] = h;
    }
    __syncthreads();

    // ---- Phase 4: d1 = relu([m1|c1] @ d1_w.T + b), (30,288)@(288,128) ----
    if (wave < 16) {
      const int rt = wave >> 3, ct = wave & 7;
      v8f acc = {};
      for (int kt = 0; kt < 288; kt += 32) {
        v16h a = load_a_f16(&dA[rt * 16][0], DA_S, kt, lane);
        v16h b = load_b_w(A.w_d1, 288, ct * 16, kt, lane);
        acc = WMMA_F16(a, b, acc);
      }
      const int col = ct * 16 + (lane & 15);
      const float bb = A.d1_b[col];
#pragma unroll
      for (int v = 0; v < 8; ++v) {
        const int b = rt * 16 + v + ((lane & 16) ? 8 : 0);
        if (b < kNB) h1s[b][col] = (_Float16)fmaxf(acc[v] + bb, 0.0f);
      }
    }
    __syncthreads();

    // ---- Phase 5: d2, log-softmax NLL, state update ----
    if (tid < kNB * 2) {
      const int b = tid >> 1, c = tid & 1;
      float s = A.d2_b[c];
      const float* w2 = A.d2_w + c * 128;
      for (int k = 0; k < 128; ++k) s += (float)h1s[b][k] * w2[k];
      s_log[b][c] = s;
      A.out[1 + (size_t)(t * kNB + b) * 2 + c] = s;
    }
    __syncthreads();
    if (tid < kNB) {
      if (s_act[tid]) {
        const float l0 = s_log[tid][0], l1 = s_log[tid][1];
        const float mx = fmaxf(l0, l1);
        const float lse = mx + logf(expf(l0 - mx) + expf(l1 - mx));
        const float nll = lse - ((s_lab[tid] == 0) ? l0 : l1);
        atomicAdd(&s_loss, nll);
      }
      prev_ids[tid] = s_ids[tid];
      prev_act[tid] = s_act[tid];
    }
    __threadfence();   // make Hf/Hm/Hc_next globally visible for next step
    __syncthreads();
  }
  if (tid == 0) A.out[0] = s_loss;
}

// ---------------------------------------------------------------------------
extern "C" void kernel_launch(void* const* d_in, const int* in_sizes, int n_in,
                              void* d_out, int out_size, void* d_ws, size_t ws_size,
                              hipStream_t stream) {
  const float* x      = (const float*)d_in[0];
  const float* y      = (const float*)d_in[1];
  const float* flow   = (const float*)d_in[2];
  const float* phi_w  = (const float*)d_in[3];
  const float* phi_b  = (const float*)d_in[4];
  const float* m_wih0 = (const float*)d_in[5];
  const float* m_whh0 = (const float*)d_in[6];
  const float* m_bih0 = (const float*)d_in[7];
  const float* m_bhh0 = (const float*)d_in[8];
  const float* m_wih1 = (const float*)d_in[9];
  const float* m_whh1 = (const float*)d_in[10];
  const float* m_bih1 = (const float*)d_in[11];
  const float* m_bhh1 = (const float*)d_in[12];
  const float* f_wih0 = (const float*)d_in[13];
  const float* f_whh0 = (const float*)d_in[14];
  const float* f_bih0 = (const float*)d_in[15];
  const float* f_bhh0 = (const float*)d_in[16];
  const float* f_wih1 = (const float*)d_in[17];
  const float* f_whh1 = (const float*)d_in[18];
  const float* f_bih1 = (const float*)d_in[19];
  const float* f_bhh1 = (const float*)d_in[20];
  const float* c_wih  = (const float*)d_in[21];
  const float* c_whh  = (const float*)d_in[22];
  const float* c_bih  = (const float*)d_in[23];
  const float* c_bhh  = (const float*)d_in[24];
  const float* d1_w   = (const float*)d_in[25];
  const float* d1_b   = (const float*)d_in[26];
  const float* d2_w   = (const float*)d_in[27];
  const float* d2_b   = (const float*)d_in[28];
  float* out = (float*)d_out;

  char* w = (char*)d_ws;
  size_t off = 0;
  auto take = [&](size_t bytes) -> void* {
    void* p = w + off;
    off = (off + bytes + 255) & ~(size_t)255;
    return p;
  };
  _Float16* phi_w16 = (_Float16*)take((size_t)kH * kXD * 2);
  _Float16* mw_ih0  = (_Float16*)take((size_t)kN3 * 2 * kH * 2);
  _Float16* mw_hh0  = (_Float16*)take((size_t)kN3 * kH * 2);
  _Float16* mw_ih1  = (_Float16*)take((size_t)kN3 * kH * 2);
  _Float16* mw_hh1  = (_Float16*)take((size_t)kN3 * kH * 2);
  _Float16* fw_ih0  = (_Float16*)take((size_t)kN3 * 2 * kH * 2);
  _Float16* fw_hh0  = (_Float16*)take((size_t)kN3 * kH * 2);
  _Float16* fw_ih1  = (_Float16*)take((size_t)kN3 * kH * 2);
  _Float16* fw_hh1  = (_Float16*)take((size_t)kN3 * kH * 2);
  _Float16* d1_w16  = (_Float16*)take((size_t)128 * 288 * 2);
  _Float16* phiX    = (_Float16*)take((size_t)kPhiRows * kH * 2);
  _Float16* phiD    = (_Float16*)take((size_t)kPhiRows * kH * 2);
  float* gi_m0 = (float*)take((size_t)kGiRows * kN3 * 4);
  float* gi_f0 = (float*)take((size_t)kGiRows * kN3 * 4);
  float* Hf = (float*)take((size_t)2 * kNB * 2 * kH * 4);
  float* Hm = (float*)take((size_t)2 * kNB * 2 * kH * 4);
  float* Hc = (float*)take((size_t)2 * kNB * kHC * 4);

  auto cvt = [&](const float* src, _Float16* dst, int n) {
    cvt_f16_kernel<<<(n + 255) / 256, 256, 0, stream>>>(src, dst, n);
  };
  cvt(phi_w,  phi_w16, kH * kXD);
  cvt(m_wih0, mw_ih0,  kN3 * 2 * kH);
  cvt(m_whh0, mw_hh0,  kN3 * kH);
  cvt(m_wih1, mw_ih1,  kN3 * kH);
  cvt(m_whh1, mw_hh1,  kN3 * kH);
  cvt(f_wih0, fw_ih0,  kN3 * 2 * kH);
  cvt(f_whh0, fw_hh0,  kN3 * kH);
  cvt(f_wih1, fw_ih1,  kN3 * kH);
  cvt(f_whh1, fw_hh1,  kN3 * kH);
  cvt(d1_w,   d1_w16,  128 * 288);

  dim3 gphi((kPhiRows + 15) / 16, kH / 128);
  phi_kernel<<<gphi, 128, 0, stream>>>(x,    phi_w16, phi_b, phiX);
  phi_kernel<<<gphi, 128, 0, stream>>>(flow, phi_w16, phi_b, phiD);

  dim3 ggi((kGiRows + 15) / 16, kN3 / 64);
  gi0_kernel<<<ggi, 128, 0, stream>>>(phiX, mw_ih0, m_bih0, gi_m0);
  gi0_kernel<<<ggi, 128, 0, stream>>>(phiD, fw_ih0, f_bih0, gi_f0);

  out_meta_kernel<<<(kT * kNB + 255) / 256, 256, 0, stream>>>(
      y, out + 1 + (size_t)kT * kNB * 2, out + 1 + (size_t)kT * kNB * 2 + kT * kNB);

  SeqArgs A;
  A.y = y; A.gi_f0 = gi_f0; A.gi_m0 = gi_m0;
  A.w_f_whh0 = fw_hh0; A.w_f_wih1 = fw_ih1; A.w_f_whh1 = fw_hh1;
  A.w_m_whh0 = mw_hh0; A.w_m_wih1 = mw_ih1; A.w_m_whh1 = mw_hh1;
  A.w_d1 = d1_w16;
  A.f_bhh0 = f_bhh0; A.f_bih1 = f_bih1; A.f_bhh1 = f_bhh1;
  A.m_bhh0 = m_bhh0; A.m_bih1 = m_bih1; A.m_bhh1 = m_bhh1;
  A.cor_wih = c_wih; A.cor_whh = c_whh; A.cor_bih = c_bih; A.cor_bhh = c_bhh;
  A.d1_b = d1_b; A.d2_w = d2_w; A.d2_b = d2_b;
  A.Hf = Hf; A.Hm = Hm; A.Hc = Hc; A.out = out;
  seq_kernel<<<1, SEQ_THREADS, 0, stream>>>(A);

  (void)in_sizes; (void)n_in; (void)out_size; (void)ws_size;
}
